// GINModel_26723286516466
// MI455X (gfx1250) — matrix operand order repaired and checked by reading, not compiled, not run on hardware
//
#include <hip/hip_runtime.h>

// ---------------- problem constants (match reference) ----------------
constexpr int NN = 50000;   // nodes
constexpr int EE = 400000;  // edges
constexpr int GG = 128;     // graphs
constexpr int DD = 256;     // hidden dim
constexpr int OUTF = 128;   // output features
constexpr float GN_EPS_C = 1e-5f;

typedef __attribute__((ext_vector_type(8)))  float  v8f;
typedef __attribute__((ext_vector_type(4)))  float  f32x4;
typedef __bf16 bf16_t;
typedef __attribute__((ext_vector_type(16))) __bf16 v16bf;
typedef __attribute__((ext_vector_type(8)))  __bf16 v8bf;
typedef __attribute__((ext_vector_type(4)))  __bf16 v4bf;

// ---------------- small utility kernels ----------------
__global__ void k_fill_zero(float* __restrict__ p, size_t n) {
  size_t i = (size_t)blockIdx.x * blockDim.x + threadIdx.x;
  if (i < n) p[i] = 0.0f;
}

__global__ void k_count(const int* __restrict__ batch, float* __restrict__ cnt, int n) {
  int i = blockIdx.x * blockDim.x + threadIdx.x;
  if (i < n) atomicAdd(&cnt[batch[i]], 1.0f);
}

// W is [256][256] row-major (k-major). WT becomes [n][k] bf16 (transposed).
__global__ void k_convert_wT(const float* __restrict__ W, bf16_t* __restrict__ WT) {
  int idx = blockIdx.x * blockDim.x + threadIdx.x;
  if (idx >= DD * DD) return;
  int n = idx >> 8, k = idx & 255;
  WT[(size_t)n * DD + k] = (bf16_t)W[(size_t)k * DD + n];
}

// node encoder: h0[n,d] = enc_b[d] + sum_{k<3} x[n,k]*enc_w[k,d]
__global__ void k_encoder(const float* __restrict__ x, const float* __restrict__ ew,
                          const float* __restrict__ eb, float* __restrict__ h0) {
  size_t idx = (size_t)blockIdx.x * blockDim.x + threadIdx.x;
  if (idx >= (size_t)NN * DD) return;
  size_t n = idx >> 8;
  int d = (int)(idx & 255);
  float v = eb[d] + x[n * 3 + 0] * ew[d] + x[n * 3 + 1] * ew[DD + d] + x[n * 3 + 2] * ew[2 * DD + d];
  h0[idx] = v;
}

// GINE message + scatter: one wave per edge, lane handles 8 consecutive d.
__global__ void k_edge_scatter(const float* __restrict__ hprev, const float* __restrict__ eattr,
                               const int* __restrict__ ei, const float* __restrict__ W,
                               const float* __restrict__ bvec, float* __restrict__ agg) {
  int t = blockIdx.x * blockDim.x + threadIdx.x;
  int e = t >> 5;
  if (e >= EE) return;
  int lane = t & 31;
  int d0 = lane << 3;
  int src = ei[e];
  int dst = ei[EE + e];
  float a0 = eattr[(size_t)e * 4 + 0];
  float a1 = eattr[(size_t)e * 4 + 1];
  float a2 = eattr[(size_t)e * 4 + 2];
  float a3 = eattr[(size_t)e * 4 + 3];
  const float* hs = hprev + (size_t)src * DD + d0;
  float* ag = agg + (size_t)dst * DD + d0;
#pragma unroll
  for (int j = 0; j < 8; ++j) {
    int d = d0 + j;
    float emb = bvec[d] + a0 * W[d] + a1 * W[DD + d] + a2 * W[2 * DD + d] + a3 * W[3 * DD + d];
    float m = hs[j] + emb;
    m = fmaxf(m, 0.0f);
    atomicAdd(ag + j, m);
  }
}

// agg += hprev   (forms z_in = (1+eps)*h + agg with eps==0)
__global__ void k_add_inplace(float* __restrict__ acc, const float* __restrict__ h, size_t n) {
  size_t i = (size_t)blockIdx.x * blockDim.x + threadIdx.x;
  if (i < n) acc[i] += h[i];
}

// ---------------- WMMA GEMM: C[M,256] = act(A[M,256] @ W[256,256] + bias) ----------------
// A: f32 row-major, staged per-block into LDS as bf16 (32 rows x 256 k, padded stride).
// BT: bf16 [n][k] (pre-transposed weights, L2-hot). act: 0=none 1=relu 2=tanh.
// Block = 256 threads = 8 waves; each wave computes a 16x64 strip (4 WMMA tiles, shared A frag).
constexpr int TILE_M = 32;
constexpr int LDS_STRIDE = DD + 8;   // +8 bf16 => 528B rows: kills 16-way LDS bank conflicts

__device__ inline void store_tile(v8f acc, float* __restrict__ C, int row0, int col,
                                  const float* __restrict__ bias, int act) {
  float b = bias[col];
  float* p = C + (size_t)row0 * DD + col;
#pragma unroll
  for (int i = 0; i < 8; ++i) {
    float v = acc[i] + b;
    if (act == 1) v = fmaxf(v, 0.0f);
    else if (act == 2) v = tanhf(v);
    p[(size_t)i * DD] = v;
  }
}

__global__ void __launch_bounds__(256)
k_gemm_wmma(const float* __restrict__ A, const bf16_t* __restrict__ BT,
            const float* __restrict__ bias, float* __restrict__ C,
            int M, int act) {
  __shared__ __align__(16) bf16_t As[TILE_M * LDS_STRIDE];
  const int tid = threadIdx.x;
  const int m_base = blockIdx.x * TILE_M;

  // ---- stage: 32x256 f32 -> bf16 LDS (each thread: 8 float4 loads, 8 b64 LDS stores) ----
#pragma unroll
  for (int it = 0; it < 8; ++it) {
    int vidx = it * 256 + tid;            // float4 index within the 32x256 tile
    int row  = vidx >> 6;                 // 64 float4 per row
    int col4 = (vidx & 63) << 2;          // k start
    int grow = m_base + row;
    if (grow > M - 1) grow = M - 1;       // clamp (tail block), no OOB reads
    f32x4 v = *(const f32x4*)(A + (size_t)grow * DD + col4);
    v4bf pk;
    pk[0] = (bf16_t)v[0]; pk[1] = (bf16_t)v[1]; pk[2] = (bf16_t)v[2]; pk[3] = (bf16_t)v[3];
    *(v4bf*)(As + row * LDS_STRIDE + col4) = pk;
  }
  __syncthreads();

  const int wid  = tid >> 5;
  const int lane = tid & 31;
  const int mt_local = wid >> 2;          // 0..1
  const int ng = wid & 3;                 // 0..3 (64-column groups)
  const int m0g = m_base + mt_local * 16;
  if (m0g >= M) return;                   // tail: wave-uniform exit, EXEC all-1s for WMMA

  const int half  = lane >> 4;            // ISA 16-bit A/B lane-half split
  const int r     = lane & 15;
  const int n0    = ng << 6;
  const int kaoff = half ? 16 : 0;        // A: 16 contiguous k per lane
  const int kboff = half ? 16 : 0;        // B: 16 contiguous k per lane
  const bf16_t* asrow = As + (mt_local * 16 + r) * LDS_STRIDE;
  const bf16_t* bbase = BT + (size_t)(n0 + r) * DD + kboff;

  v8f acc0 = {}, acc1 = {}, acc2 = {}, acc3 = {};
#pragma unroll
  for (int kk = 0; kk < DD; kk += 32) {
    union { v8bf h[2]; v16bf v; } af;
    af.h[0] = *(const v8bf*)(asrow + kk + kaoff);
    af.h[1] = *(const v8bf*)(asrow + kk + kaoff + 8);
    v16bf b0 = *(const v16bf*)(bbase + kk);
    v16bf b1 = *(const v16bf*)(bbase + (size_t)16 * DD + kk);
    v16bf b2 = *(const v16bf*)(bbase + (size_t)32 * DD + kk);
    v16bf b3 = *(const v16bf*)(bbase + (size_t)48 * DD + kk);
    acc0 = __builtin_amdgcn_wmma_f32_16x16x32_bf16(false, af.v, false, b0, (short)0, acc0, false, false);
    acc1 = __builtin_amdgcn_wmma_f32_16x16x32_bf16(false, af.v, false, b1, (short)0, acc1, false, false);
    acc2 = __builtin_amdgcn_wmma_f32_16x16x32_bf16(false, af.v, false, b2, (short)0, acc2, false, false);
    acc3 = __builtin_amdgcn_wmma_f32_16x16x32_bf16(false, af.v, false, b3, (short)0, acc3, false, false);
  }
  const int row0 = m0g + half * 8;
  store_tile(acc0, C, row0, n0 +  0 + r, bias, act);
  store_tile(acc1, C, row0, n0 + 16 + r, bias, act);
  store_tile(acc2, C, row0, n0 + 32 + r, bias, act);
  store_tile(acc3, C, row0, n0 + 48 + r, bias, act);
}

// ---------------- GraphNorm (two-pass segment stats via L2-resident atomics) ----------------
__global__ void k_stats_sum(const float* __restrict__ z, const int* __restrict__ batch,
                            float* __restrict__ sums) {
  size_t idx = (size_t)blockIdx.x * blockDim.x + threadIdx.x;
  if (idx >= (size_t)NN * DD) return;
  int n = (int)(idx >> 8);
  int d = (int)(idx & 255);
  atomicAdd(&sums[(size_t)batch[n] * DD + d], z[idx]);
}

__global__ void k_stats_var(const float* __restrict__ z, const int* __restrict__ batch,
                            const float* __restrict__ sums, const float* __restrict__ cnt,
                            const float* __restrict__ ms, float* __restrict__ var) {
  size_t idx = (size_t)blockIdx.x * blockDim.x + threadIdx.x;
  if (idx >= (size_t)NN * DD) return;
  int n = (int)(idx >> 8);
  int d = (int)(idx & 255);
  int g = batch[n];
  float c = fmaxf(cnt[g], 1.0f);
  float mean = sums[(size_t)g * DD + d] / c;
  float sub = z[idx] - mean * ms[d];
  atomicAdd(&var[(size_t)g * DD + d], sub * sub);
}

__global__ void k_norm_res(const float* __restrict__ z, const float* __restrict__ hprev,
                           const int* __restrict__ batch, const float* __restrict__ sums,
                           const float* __restrict__ var, const float* __restrict__ cnt,
                           const float* __restrict__ gw, const float* __restrict__ gb,
                           const float* __restrict__ ms, float* __restrict__ hout) {
  size_t idx = (size_t)blockIdx.x * blockDim.x + threadIdx.x;
  if (idx >= (size_t)NN * DD) return;
  int n = (int)(idx >> 8);
  int d = (int)(idx & 255);
  int g = batch[n];
  float c = fmaxf(cnt[g], 1.0f);
  float mean = sums[(size_t)g * DD + d] / c;
  float sub = z[idx] - mean * ms[d];
  float va = var[(size_t)g * DD + d] / c;
  float v = gw[d] * sub * rsqrtf(va + GN_EPS_C) + gb[d];
  hout[idx] = v + hprev[idx];
}

// ---------------- JK attention ----------------
// scores[n,l] = dot(S[n,:], attn_w2)
__global__ void k_attn_dot(const float* __restrict__ S, const float* __restrict__ w2v,
                           float* __restrict__ scores, int l) {
  int n = blockIdx.x * blockDim.x + threadIdx.x;
  if (n >= NN) return;
  const float* row = S + (size_t)n * DD;
  float s = 0.0f;
#pragma unroll 4
  for (int d = 0; d < DD; ++d) s += row[d] * w2v[d];
  scores[(size_t)n * 3 + l] = s;
}

// softmax over L=3 + weighted sum + global_add_pool (one wave per node)
__global__ void k_jk_pool(const float* __restrict__ hb0, const float* __restrict__ hb1,
                          const float* __restrict__ hb2, const float* __restrict__ scores,
                          const int* __restrict__ batch, float* __restrict__ pooled) {
  int t = blockIdx.x * blockDim.x + threadIdx.x;
  int n = t >> 5;
  if (n >= NN) return;
  int lane = t & 31;
  float s0 = scores[(size_t)n * 3 + 0];
  float s1 = scores[(size_t)n * 3 + 1];
  float s2 = scores[(size_t)n * 3 + 2];
  float mx = fmaxf(s0, fmaxf(s1, s2));
  float e0 = expf(s0 - mx), e1 = expf(s1 - mx), e2 = expf(s2 - mx);
  float inv = 1.0f / (e0 + e1 + e2);
  e0 *= inv; e1 *= inv; e2 *= inv;
  int g = batch[n];
  size_t base = (size_t)n * DD + lane * 8;
  float* pg = pooled + (size_t)g * DD + lane * 8;
#pragma unroll
  for (int j = 0; j < 8; ++j) {
    float v = e0 * hb0[base + j] + e1 * hb1[base + j] + e2 * hb2[base + j];
    atomicAdd(pg + j, v);
  }
}

// final FC: out[g,o] = fc_b[o] + sum_d pooled[g,d]*fc_w[d,o]
__global__ void k_fc(const float* __restrict__ pooled, const float* __restrict__ fw,
                     const float* __restrict__ fb, float* __restrict__ out) {
  int t = blockIdx.x * blockDim.x + threadIdx.x;
  if (t >= GG * OUTF) return;
  int g = t / OUTF, o = t % OUTF;
  const float* pr = pooled + (size_t)g * DD;
  float s = fb[o];
#pragma unroll 4
  for (int d = 0; d < DD; ++d) s += pr[d] * fw[(size_t)d * OUTF + o];
  out[t] = s;
}

// ---------------- host launcher ----------------
extern "C" void kernel_launch(void* const* d_in, const int* in_sizes, int n_in,
                              void* d_out, int out_size, void* d_ws, size_t ws_size,
                              hipStream_t stream) {
  (void)in_sizes; (void)n_in; (void)out_size; (void)ws_size;
  const float* x       = (const float*)d_in[0];
  const int*   ei      = (const int*)d_in[1];
  const float* eattr   = (const float*)d_in[2];
  const int*   batch   = (const int*)d_in[3];
  const float* enc_w   = (const float*)d_in[4];
  const float* enc_b   = (const float*)d_in[5];
  const float* edge_w  = (const float*)d_in[6];
  const float* edge_b  = (const float*)d_in[7];
  const float* w1      = (const float*)d_in[8];
  const float* b1      = (const float*)d_in[9];
  const float* w2      = (const float*)d_in[10];
  const float* b2      = (const float*)d_in[11];
  const float* gn_w    = (const float*)d_in[12];
  const float* gn_b    = (const float*)d_in[13];
  const float* gn_ms   = (const float*)d_in[14];
  const float* attn_w1 = (const float*)d_in[15];
  const float* attn_b1 = (const float*)d_in[16];
  const float* attn_w2 = (const float*)d_in[17];
  const float* fc_w    = (const float*)d_in[18];
  const float* fc_b    = (const float*)d_in[19];
  float* out = (float*)d_out;

  char* ws = (char*)d_ws;
  size_t off = 0;
  auto take = [&](size_t bytes) -> char* {
    char* p = ws + off;
    off = (off + bytes + 255) & ~(size_t)255;
    return p;
  };
  const size_t ND = (size_t)NN * DD;
  float* h0   = (float*)take(ND * 4);
  float* hb0  = (float*)take(ND * 4);
  float* hb1  = (float*)take(ND * 4);
  float* hb2  = (float*)take(ND * 4);
  float* hbv[3] = {hb0, hb1, hb2};
  float* agg  = (float*)take(ND * 4);
  float* tbuf = (float*)take(ND * 4);
  float* zbuf = (float*)take(ND * 4);
  float* sums = (float*)take((size_t)GG * DD * 4);
  float* var  = (float*)take((size_t)GG * DD * 4);
  float* cnt  = (float*)take((size_t)GG * 4);
  float* scor = (float*)take((size_t)NN * 3 * 4);
  float* pool = (float*)take((size_t)GG * DD * 4);
  bf16_t* w1T = (bf16_t*)take((size_t)3 * DD * DD * 2);
  bf16_t* w2T = (bf16_t*)take((size_t)3 * DD * DD * 2);
  bf16_t* awT = (bf16_t*)take((size_t)DD * DD * 2);

  const dim3 blk(256);
  const int bND = (int)((ND + 255) / 256);
  const int bGD = (GG * DD + 255) / 256;
  const int gemm_blocks = (NN + TILE_M - 1) / TILE_M;

  // counts + zero pooled
  k_fill_zero<<<bGD, blk, 0, stream>>>(pool, (size_t)GG * DD);
  k_fill_zero<<<1, blk, 0, stream>>>(cnt, (size_t)GG);
  k_count<<<(NN + 255) / 256, blk, 0, stream>>>(batch, cnt, NN);

  // pre-transpose weights to bf16 [n][k]
  for (int i = 0; i < 3; ++i) {
    k_convert_wT<<<DD * DD / 256, blk, 0, stream>>>(w1 + (size_t)i * DD * DD, w1T + (size_t)i * DD * DD);
    k_convert_wT<<<DD * DD / 256, blk, 0, stream>>>(w2 + (size_t)i * DD * DD, w2T + (size_t)i * DD * DD);
  }
  k_convert_wT<<<DD * DD / 256, blk, 0, stream>>>(attn_w1, awT);

  // node encoder
  k_encoder<<<bND, blk, 0, stream>>>(x, enc_w, enc_b, h0);

  const float* hprev = h0;
  for (int i = 0; i < 3; ++i) {
    k_fill_zero<<<bND, blk, 0, stream>>>(agg, ND);
    k_edge_scatter<<<(EE * 32 + 255) / 256, blk, 0, stream>>>(
        hprev, eattr, ei, edge_w + (size_t)i * 4 * DD, edge_b + (size_t)i * DD, agg);
    k_add_inplace<<<bND, blk, 0, stream>>>(agg, hprev, ND);

    k_gemm_wmma<<<gemm_blocks, blk, 0, stream>>>(agg,  w1T + (size_t)i * DD * DD, b1 + (size_t)i * DD, tbuf, NN, 1);
    k_gemm_wmma<<<gemm_blocks, blk, 0, stream>>>(tbuf, w2T + (size_t)i * DD * DD, b2 + (size_t)i * DD, zbuf, NN, 1);

    k_fill_zero<<<bGD, blk, 0, stream>>>(sums, (size_t)GG * DD);
    k_fill_zero<<<bGD, blk, 0, stream>>>(var, (size_t)GG * DD);
    k_stats_sum<<<bND, blk, 0, stream>>>(zbuf, batch, sums);
    k_stats_var<<<bND, blk, 0, stream>>>(zbuf, batch, sums, cnt, gn_ms + (size_t)i * DD, var);
    k_norm_res<<<bND, blk, 0, stream>>>(zbuf, hprev, batch, sums, var, cnt,
                                        gn_w + (size_t)i * DD, gn_b + (size_t)i * DD,
                                        gn_ms + (size_t)i * DD, hbv[i]);
    hprev = hbv[i];
  }

  // JK attention: scores per layer, then softmax-weighted pooling
  for (int l = 0; l < 3; ++l) {
    k_gemm_wmma<<<gemm_blocks, blk, 0, stream>>>(hbv[l], awT, attn_b1, tbuf, NN, 2);
    k_attn_dot<<<(NN + 255) / 256, blk, 0, stream>>>(tbuf, attn_w2, scor, l);
  }
  k_jk_pool<<<(NN * 32 + 255) / 256, blk, 0, stream>>>(hb0, hb1, hb2, scor, batch, pool);
  k_fc<<<(GG * OUTF + 255) / 256, blk, 0, stream>>>(pool, fc_w, fc_b, out);
}
